// SwinBlock_16716012716557
// MI455X (gfx1250) — matrix operand order repaired and checked by reading, not compiled, not run on hardware
//
#include <hip/hip_runtime.h>
#include <hip/hip_bf16.h>
#include <stdint.h>

typedef __bf16 bf16;
typedef __attribute__((ext_vector_type(16))) __bf16   v16bf;
typedef __attribute__((ext_vector_type(8)))  __bf16   v8bf;
typedef __attribute__((ext_vector_type(8)))  float    v8f;
typedef __attribute__((ext_vector_type(4)))  uint32_t v4u;

#define DEV static __device__ __forceinline__

constexpr int   kB     = 64;
constexpr int   kH     = 56;
constexpr int   kW     = 56;
constexpr int   kC     = 192;
constexpr int   kHeads = 6;
constexpr int   kShift = 4;
constexpr int   kNW    = 49;          // windows per image (7x7)
constexpr int   kL     = kH * kW;     // 3136
constexpr int   kRows  = kB * kL;     // 200704 tokens
constexpr int   kHid   = 768;
constexpr float kScale = 0.17677669529663687f;  // 1/sqrt(32)

union V16 { v16bf v; v8bf h[2]; };
union BPU { v16bf v; v4u  u[2]; };
union PKU { uint32_t u; __bf16 h[2]; };

DEV v16bf make_frag(v8bf lo, v8bf hi) { V16 t; t.h[0] = lo; t.h[1] = hi; return t.v; }

DEV v8f wmma_bf16(v16bf a, v16bf b, v8f c) {
  return __builtin_amdgcn_wmma_f32_16x16x32_bf16(false, a, false, b, (short)0, c, false, false);
}

DEV float gelu_exact(float x) { return 0.5f * x * (1.0f + erff(x * 0.70710678118654752f)); }

DEV int region_of(int wi, int tok) {
  int gh = (wi / 7) * 8 + (tok >> 3);
  int gw = (wi % 7) * 8 + (tok & 7);
  int rh = gh < 48 ? 0 : (gh < 52 ? 1 : 2);
  int rw = gw < 48 ? 0 : (gw < 52 ? 1 : 2);
  return rh * 3 + rw;
}

// -------- weight pre-pack into WMMA B-fragment order --------
// dst[((kt*(N/16)+nt)*32 + lane)*8 + j] = pack(W[kt*32+g*16+2j][nt*16+n], W[..2j+1..])
__global__ __launch_bounds__(256) void pack_weights(const float* __restrict__ Wsrc,
                                                    uint32_t* __restrict__ dst,
                                                    int K, int N) {
  int idx = blockIdx.x * blockDim.x + threadIdx.x;
  int total = (K / 32) * (N / 16) * 256;
  if (idx >= total) return;
  int j    = idx & 7;
  int lane = (idx >> 3) & 31;
  int tile = idx >> 8;
  int nt   = tile % (N / 16);
  int kt   = tile / (N / 16);
  int g = lane >> 4, n = lane & 15;
  int k0  = kt * 32 + g * 16 + 2 * j;
  int col = nt * 16 + n;
  PKU p;
  p.h[0] = (bf16)Wsrc[(size_t)k0 * N + col];
  p.h[1] = (bf16)Wsrc[(size_t)(k0 + 1) * N + col];
  dst[idx] = p.u;
}

// -------- relative position bias table: bias[h][n][m] --------
__global__ __launch_bounds__(256) void build_bias(const float* __restrict__ rpb,
                                                  float* __restrict__ bias) {
  int idx = blockIdx.x * blockDim.x + threadIdx.x;
  if (idx >= kHeads * 64 * 64) return;
  int m = idx & 63, n = (idx >> 6) & 63, h = idx >> 12;
  int r1 = n >> 3, c1 = n & 7, r2 = m >> 3, c2 = m & 7;
  int rp = (r1 - r2 + 7) * 15 + (c1 - c2 + 7);
  bias[idx] = rpb[rp * kHeads + h];
}

// -------- LayerNorm (+ optional roll/window-partition remap), fp32 -> bf16 --------
__global__ __launch_bounds__(256) void ln_rows(const float* __restrict__ x,
                                               const float* __restrict__ w,
                                               const float* __restrict__ b,
                                               bf16* __restrict__ out, int remap) {
  int row = blockIdx.x * 8 + (threadIdx.x >> 5);
  if (row >= kRows) return;
  int lane = threadIdx.x & 31;
  size_t src_row;
  if (remap) {  // row is in window layout; read from rolled source position
    int wdx = row >> 6, n = row & 63;
    int bb = wdx / kNW, wi = wdx % kNW;
    int gh = (wi / 7) * 8 + (n >> 3);
    int gw = (wi % 7) * 8 + (n & 7);
    int hs = (gh + kShift) % kH;
    int ws_ = (gw + kShift) % kW;
    src_row = (size_t)bb * kL + hs * kW + ws_;
  } else {
    src_row = (size_t)row;
  }
  const float* src = x + src_row * (size_t)kC;
  float v[6];
  float s = 0.f;
#pragma unroll
  for (int e = 0; e < 6; ++e) { v[e] = src[e * 32 + lane]; s += v[e]; }
#pragma unroll
  for (int mk = 1; mk < 32; mk <<= 1) s += __shfl_xor(s, mk, 32);
  float mu = s * (1.f / 192.f);
  float vs = 0.f;
#pragma unroll
  for (int e = 0; e < 6; ++e) { float d = v[e] - mu; vs += d * d; }
#pragma unroll
  for (int mk = 1; mk < 32; mk <<= 1) vs += __shfl_xor(vs, mk, 32);
  float inv = rsqrtf(vs * (1.f / 192.f) + 1e-5f);
  bf16* dst = out + (size_t)row * kC;
#pragma unroll
  for (int e = 0; e < 6; ++e)
    dst[e * 32 + lane] = (bf16)((v[e] - mu) * inv * w[e * 32 + lane] + b[e * 32 + lane]);
}

// -------- generic bf16 WMMA GEMM: one wave computes a 16x64 strip --------
// mode 0: QKV  -> scatter q[w][h][tok][dd], k[w][h][tok][dd], vT[w][h][dd][tok]
// mode 1: proj -> window-reverse + roll + residual, fp32 out
// mode 2: FC1  -> GELU, bf16 out
// mode 3: FC2  -> + residual, fp32 out (d_out)
__global__ __launch_bounds__(256) void gemm_wmma(
    const bf16* __restrict__ A, const uint32_t* __restrict__ Bp,
    const float* __restrict__ bias, int Kdim, int Ndim, int mode,
    bf16* __restrict__ outb, float* __restrict__ outf,
    const float* __restrict__ addsrc,
    bf16* __restrict__ qb, bf16* __restrict__ kb, bf16* __restrict__ vtb) {
  const int lane    = threadIdx.x & 31;
  const int gwave   = blockIdx.x * 8 + (threadIdx.x >> 5);
  const int ngroups = Ndim >> 6;
  const int mt = gwave / ngroups;
  const int ng = gwave % ngroups;
  if (mt >= kRows / 16) return;
  const int m      = lane & 15;
  const int koff   = (lane >> 4) << 4;
  const int nt_tot = Ndim >> 4;

  const v8f vzero = {0.f, 0.f, 0.f, 0.f, 0.f, 0.f, 0.f, 0.f};
  v8f acc[4];
#pragma unroll
  for (int j = 0; j < 4; ++j) acc[j] = vzero;

  const bf16* arow = A + (size_t)(mt * 16 + m) * Kdim;
  for (int kt = 0; kt < (Kdim >> 5); ++kt) {
    v8bf alo = *(const v8bf*)(arow + kt * 32 + koff);
    v8bf ahi = *(const v8bf*)(arow + kt * 32 + koff + 8);
    v16bf af = make_frag(alo, ahi);
#pragma unroll
    for (int j = 0; j < 4; ++j) {
      const uint32_t* bp = Bp + (((size_t)kt * nt_tot + (ng * 4 + j)) * 32 + lane) * 8;
      BPU bb;
      bb.u[0] = *(const v4u*)bp;
      bb.u[1] = *(const v4u*)(bp + 4);
      acc[j] = wmma_bf16(af, bb.v, acc[j]);
    }
  }

  const int ro_base = (lane >> 4) << 3;  // C-layout: lanes 16-31 hold rows M+8
  const int ncol    = lane & 15;
#pragma unroll
  for (int j = 0; j < 4; ++j) {
    int col = (ng * 4 + j) * 16 + ncol;
    float bv = bias[col];
#pragma unroll
    for (int r = 0; r < 8; ++r) {
      int gr = mt * 16 + ro_base + r;
      float val = acc[j][r] + bv;
      if (mode == 0) {
        int wdx = gr >> 6, tok = gr & 63;
        int t  = col / 192;
        int rm = col % 192;
        int h  = rm >> 5;
        int dd = rm & 31;
        size_t hb = ((size_t)wdx * kHeads + h) * 2048;
        if (t == 0)      qb [hb + (size_t)tok * 32 + dd] = (bf16)val;
        else if (t == 1) kb [hb + (size_t)tok * 32 + dd] = (bf16)val;
        else             vtb[hb + (size_t)dd * 64 + tok] = (bf16)val;   // V transposed
      } else if (mode == 1) {
        int wdx = gr >> 6, n = gr & 63;
        int bb2 = wdx / kNW, wi = wdx % kNW;
        int gh = (wi / 7) * 8 + (n >> 3);
        int gw = (wi % 7) * 8 + (n & 7);
        int hd = (gh + kShift) % kH;
        int wd = (gw + kShift) % kW;
        size_t drow = (size_t)bb2 * kL + hd * kW + wd;
        outf[drow * kC + col] = val + addsrc[drow * kC + col];
      } else if (mode == 2) {
        outb[(size_t)gr * kHid + col] = (bf16)gelu_exact(val);
      } else {
        outf[(size_t)gr * kC + col] = val + addsrc[(size_t)gr * kC + col];
      }
    }
  }
}

// -------- windowed attention: one (window, head) per wave --------
__global__ __launch_bounds__(128) void attn_wmma(
    const bf16* __restrict__ qb, const bf16* __restrict__ kb,
    const bf16* __restrict__ vtb, const float* __restrict__ biasf,
    bf16* __restrict__ outb) {
  __shared__ alignas(16) bf16 plds_all[4][64 * 72];  // padded: stride 72 = 144B
  const int wslot = threadIdx.x >> 5;
  const int lane  = threadIdx.x & 31;
  const int task  = blockIdx.x * 4 + wslot;
  if (task >= 3136 * kHeads) return;
  const int wdx = task / kHeads;
  const int h   = task % kHeads;
  const int wi  = wdx % kNW;

  const size_t base = (size_t)task * 2048;  // (wdx*6 + h) * 64*32
  const bf16* Q  = qb + base;
  const bf16* Km = kb + base;
  const bf16* Vt = vtb + base;
  bf16* P = &plds_all[wslot][0];

  const int m    = lane & 15;
  const int g    = lane >> 4;
  const int koff = g << 4;
  const int n15  = lane & 15;

  // Q A-fragments: 4 row tiles of 16x32
  v16bf aq[4];
#pragma unroll
  for (int i = 0; i < 4; ++i) {
    v8bf lo = *(const v8bf*)(Q + (i * 16 + m) * 32 + koff);
    v8bf hi = *(const v8bf*)(Q + (i * 16 + m) * 32 + koff + 8);
    aq[i] = make_frag(lo, hi);
  }

  const v8f vzero = {0.f, 0.f, 0.f, 0.f, 0.f, 0.f, 0.f, 0.f};
  v8f S[4][4];
#pragma unroll
  for (int i = 0; i < 4; ++i)
#pragma unroll
    for (int j = 0; j < 4; ++j) S[i][j] = vzero;

  // S = Q @ K^T : K stored row-major [tok][dd] is exactly B^T fragment order
#pragma unroll
  for (int j = 0; j < 4; ++j) {
    v8bf lo = *(const v8bf*)(Km + (j * 16 + n15) * 32 + g * 16);
    v8bf hi = *(const v8bf*)(Km + (j * 16 + n15) * 32 + g * 16 + 8);
    v16bf bk = make_frag(lo, hi);
#pragma unroll
    for (int i = 0; i < 4; ++i) S[i][j] = wmma_bf16(aq[i], bk, S[i][j]);
  }

  // scale + relative-position bias + shift mask, then rowwise softmax
  int regk[4];
#pragma unroll
  for (int j = 0; j < 4; ++j) regk[j] = region_of(wi, j * 16 + n15);
  const float* brow = biasf + (size_t)h * 4096;

#pragma unroll
  for (int i = 0; i < 4; ++i) {
    float mx[8];
#pragma unroll
    for (int r = 0; r < 8; ++r) {
      int qn = i * 16 + g * 8 + r;
      int regq = region_of(wi, qn);
      float rowm = -1e30f;
#pragma unroll
      for (int j = 0; j < 4; ++j) {
        int km = j * 16 + n15;
        float v = S[i][j][r] * kScale + brow[qn * 64 + km] +
                  ((regq != regk[j]) ? -100.f : 0.f);
        S[i][j][r] = v;
        rowm = fmaxf(rowm, v);
      }
#pragma unroll
      for (int mk = 1; mk < 16; mk <<= 1) rowm = fmaxf(rowm, __shfl_xor(rowm, mk, 32));
      mx[r] = rowm;
    }
#pragma unroll
    for (int r = 0; r < 8; ++r) {
      float rs = 0.f;
#pragma unroll
      for (int j = 0; j < 4; ++j) {
        float e = __expf(S[i][j][r] - mx[r]);
        S[i][j][r] = e;
        rs += e;
      }
#pragma unroll
      for (int mk = 1; mk < 16; mk <<= 1) rs += __shfl_xor(rs, mk, 32);
      float rinv = 1.f / rs;
#pragma unroll
      for (int j = 0; j < 4; ++j) {
        float p = S[i][j][r] * rinv;
        P[(i * 16 + g * 8 + r) * 72 + j * 16 + n15] = (bf16)p;  // row-major P in LDS
      }
    }
  }
  asm volatile("s_wait_dscnt 0" ::: "memory");

  // O = P @ V, via V^T (row-major [dd][tok]) giving contiguous B fragments
#pragma unroll
  for (int i = 0; i < 4; ++i) {
    v16bf ap[2];
#pragma unroll
    for (int kc = 0; kc < 2; ++kc) {
      v8bf lo = *(const v8bf*)(P + (i * 16 + m) * 72 + kc * 32 + koff);
      v8bf hi = *(const v8bf*)(P + (i * 16 + m) * 72 + kc * 32 + koff + 8);
      ap[kc] = make_frag(lo, hi);
    }
#pragma unroll
    for (int jc = 0; jc < 2; ++jc) {
      v8f o = vzero;
#pragma unroll
      for (int kc = 0; kc < 2; ++kc) {
        v8bf lo = *(const v8bf*)(Vt + (jc * 16 + n15) * 64 + kc * 32 + g * 16);
        v8bf hi = *(const v8bf*)(Vt + (jc * 16 + n15) * 64 + kc * 32 + g * 16 + 8);
        o = wmma_bf16(ap[kc], make_frag(lo, hi), o);
      }
#pragma unroll
      for (int r = 0; r < 8; ++r) {
        int tok = i * 16 + g * 8 + r;
        int dd  = jc * 16 + n15;
        outb[((size_t)wdx * 64 + tok) * kC + h * 32 + dd] = (bf16)o[r];
      }
    }
  }
}

extern "C" void kernel_launch(void* const* d_in, const int* in_sizes, int n_in,
                              void* d_out, int out_size, void* d_ws, size_t ws_size,
                              hipStream_t stream) {
  (void)in_sizes; (void)n_in; (void)out_size; (void)ws_size;
  const float* x     = (const float*)d_in[0];
  const float* ln1w  = (const float*)d_in[1];
  const float* ln1b  = (const float*)d_in[2];
  const float* qkvw  = (const float*)d_in[3];
  const float* qkvb  = (const float*)d_in[4];
  const float* projw = (const float*)d_in[5];
  const float* projb = (const float*)d_in[6];
  const float* rpb   = (const float*)d_in[7];
  const float* ln2w  = (const float*)d_in[8];
  const float* ln2b  = (const float*)d_in[9];
  const float* fc1w  = (const float*)d_in[10];
  const float* fc1b  = (const float*)d_in[11];
  const float* fc2w  = (const float*)d_in[12];
  const float* fc2b  = (const float*)d_in[13];

  char* ws = (char*)d_ws;
  const size_t SZ_BF = (size_t)kRows * kC * 2;  // 77,070,336 B
  bf16*  xw    = (bf16*)(ws);                   // LN1 output; reused as attn-out; part of h1
  bf16*  qbuf  = (bf16*)(ws + 1 * SZ_BF);
  bf16*  kbuf  = (bf16*)(ws + 2 * SZ_BF);
  bf16*  vtbuf = (bf16*)(ws + 3 * SZ_BF);
  float* xres  = (float*)(ws + 4 * SZ_BF);      // kRows*kC fp32
  bf16*  x2n   = (bf16*)(ws + 4 * SZ_BF + (size_t)kRows * kC * 4);
  char*  wp    = ws + 4 * SZ_BF + (size_t)kRows * kC * 4 + SZ_BF;
  uint32_t* wq  = (uint32_t*)wp;
  uint32_t* wpr = wq + 6 * 36 * 256;
  uint32_t* w1  = wpr + 6 * 12 * 256;
  uint32_t* w2  = w1 + 6 * 48 * 256;
  float* biasf  = (float*)(w2 + 24 * 12 * 256);
  bf16*  h1     = (bf16*)(ws);                  // 308 MB, overlaps xw/q/k/vt (dead by then)

  // setup: weight packing + bias table (tiny, every call; deterministic)
  pack_weights<<<(6 * 36 * 256 + 255) / 256, 256, 0, stream>>>(qkvw, wq, 192, 576);
  pack_weights<<<(6 * 12 * 256 + 255) / 256, 256, 0, stream>>>(projw, wpr, 192, 192);
  pack_weights<<<(6 * 48 * 256 + 255) / 256, 256, 0, stream>>>(fc1w, w1, 192, 768);
  pack_weights<<<(24 * 12 * 256 + 255) / 256, 256, 0, stream>>>(fc2w, w2, 768, 192);
  build_bias<<<(kHeads * 64 * 64 + 255) / 256, 256, 0, stream>>>(rpb, biasf);

  // LN1 + roll + window partition (window-layout bf16 rows)
  ln_rows<<<kRows / 8, 256, 0, stream>>>(x, ln1w, ln1b, xw, 1);

  const int mtiles = kRows / 16;  // 12544
  auto blocks = [&](int ng) { return (mtiles * ng + 7) / 8; };

  // QKV projection (N=576), scatter q/k/v^T per window-head
  gemm_wmma<<<blocks(9), 256, 0, stream>>>(xw, wq, qkvb, 192, 576, 0,
                                           nullptr, nullptr, nullptr, qbuf, kbuf, vtbuf);
  // windowed attention, one (window, head) per wave; output reuses xw
  attn_wmma<<<(3136 * kHeads) / 4, 128, 0, stream>>>(qbuf, kbuf, vtbuf, biasf, xw);
  // proj + window reverse + roll-back + residual -> xres (fp32)
  gemm_wmma<<<blocks(3), 256, 0, stream>>>(xw, wpr, projb, 192, 192, 1,
                                           nullptr, xres, x, nullptr, nullptr, nullptr);
  // LN2 -> x2n (bf16)
  ln_rows<<<kRows / 8, 256, 0, stream>>>(xres, ln2w, ln2b, x2n, 0);
  // FC1 + GELU -> h1 (bf16)
  gemm_wmma<<<blocks(12), 256, 0, stream>>>(x2n, w1, fc1b, 192, 768, 2,
                                            h1, nullptr, nullptr, nullptr, nullptr, nullptr);
  // FC2 + residual -> d_out (fp32)
  gemm_wmma<<<blocks(3), 256, 0, stream>>>(h1, w2, fc2b, 768, 192, 3,
                                           nullptr, (float*)d_out, xres, nullptr, nullptr, nullptr);
}
